// GraphToGraph_65695819760358
// MI455X (gfx1250) — matrix operand order repaired and checked by reading, not compiled, not run on hardware
//
#include <hip/hip_runtime.h>
#include <stdint.h>

// ---------------------------------------------------------------------------
// GraphToGraph pipeline for gfx1250 (MI455X, wave32, WMMA).
// Dense GEMMs (struc2 / con2 / edge-MLP) use v_wmma_f32_16x16x32_bf16.
// Weights are staged in LDS column-major so each lane's B-fragment is a
// contiguous 32-byte run -> ds_load_b128 pairs instead of 16x ds_load_u16.
// Irregular graph work uses atomics, bitonic sorts, chunked scans.
// ---------------------------------------------------------------------------

constexpr int N_NODES = 50000;
constexpr int E_PAIRS = 250000;
constexpr int ETOT    = 550000;          // 2*E + N
constexpr int DCI     = 128;
constexpr int DSI     = 64;
constexpr int SORT_SZ = 1 << 20;         // padded sort size (>= ETOT)
constexpr unsigned NN = 50000u;

typedef __bf16 bf16;
typedef bf16  v16bf __attribute__((ext_vector_type(16)));
typedef float v8f   __attribute__((ext_vector_type(8)));
typedef unsigned short v16us __attribute__((ext_vector_type(16)));

__device__ __forceinline__ unsigned short f2bf(float f) {
  union { float f; unsigned u; } c; c.f = f;
  unsigned u = c.u;                           // round-to-nearest-even
  return (unsigned short)((u + 0x7FFFu + ((u >> 16) & 1u)) >> 16);
}
__device__ __forceinline__ v16bf as_bf(v16us v) { return __builtin_bit_cast(v16bf, v); }
__device__ __forceinline__ float frelu(float x) { return x > 0.f ? x : 0.f; }

// ---------------------------------------------------------------------------
// Edge aggregation: agg[dst] += w * feats[src]   (D = chunks*4 floats/row)
// ---------------------------------------------------------------------------
__global__ void k_agg(const float* __restrict__ feats, const int* __restrict__ eid,
                      const float* __restrict__ ew, float* __restrict__ agg,
                      int D, int chunks) {
  int idx = blockIdx.x * blockDim.x + threadIdx.x;
  int tot = ETOT * chunks;
  if (idx >= tot) return;
  int edge = idx / chunks, q = idx % chunks;
  int src = eid[edge], dst = eid[ETOT + edge];
  float w = ew[edge];
  const float* sp = feats + (size_t)src * D + q * 4;
  float* dp = agg + (size_t)dst * D + q * 4;
  atomicAdd(dp + 0, w * sp[0]); atomicAdd(dp + 1, w * sp[1]);
  atomicAdd(dp + 2, w * sp[2]); atomicAdd(dp + 3, w * sp[3]);
}

// ---------------------------------------------------------------------------
// WMMA GEMM 1: struc2 = relu([agg_s | struc_feats](N x128) @ [Ws;Us](128x64) + bs)
// One wave per 16-row strip; 4 N-tiles; K=128 in 4 chunks of 32.
// LDS weights column-major: wl[col*128 + k].
// ---------------------------------------------------------------------------
__global__ __launch_bounds__(256) void k_struc2(
    const float* __restrict__ agg, const float* __restrict__ sf,
    const float* __restrict__ Ws, const float* __restrict__ Us,
    const float* __restrict__ bs, float* __restrict__ out) {
  __shared__ unsigned short wl[64 * 128];   // [col][k]
  __shared__ float bl[64];
  for (int t = (int)threadIdx.x; t < 128 * 64; t += 256) {
    int k = t >> 6, c = t & 63;             // coalesced global read
    wl[c * 128 + k] = f2bf(k < 64 ? Ws[k * 64 + c] : Us[(k - 64) * 64 + c]);
  }
  if (threadIdx.x < 64) bl[threadIdx.x] = bs[threadIdx.x];
  __syncthreads();

  int wave = threadIdx.x >> 5, lane = threadIdx.x & 31;
  int strip = blockIdx.x * 8 + wave;
  if (strip >= N_NODES / 16) return;
  int m = lane & 15, hlf = lane >> 4;
  int row = strip * 16 + m;

  v8f zero;
#pragma unroll
  for (int e = 0; e < 8; ++e) zero[e] = 0.f;
  v8f acc[4];
#pragma unroll
  for (int t = 0; t < 4; ++t) acc[t] = zero;

#pragma unroll
  for (int kc = 0; kc < 4; ++kc) {
    int kk0 = kc * 32 + hlf * 16, kk1 = kc * 32 + hlf * 16 + 8, kkb = kc * 32 + hlf * 16;
    const float* p0 = kk0 < 64 ? agg + (size_t)row * 64 + kk0 : sf + (size_t)row * 64 + (kk0 - 64);
    const float* p1 = kk1 < 64 ? agg + (size_t)row * 64 + kk1 : sf + (size_t)row * 64 + (kk1 - 64);
    v16us a;
#pragma unroll
    for (int e = 0; e < 8; ++e) { a[e] = f2bf(p0[e]); a[8 + e] = f2bf(p1[e]); }
#pragma unroll
    for (int nt = 0; nt < 4; ++nt) {
      int col = nt * 16 + m;
      v16us b = *reinterpret_cast<const v16us*>(wl + (size_t)col * 128 + kkb);
      acc[nt] = __builtin_amdgcn_wmma_f32_16x16x32_bf16(
          false, as_bf(a), false, as_bf(b), (short)0, acc[nt], false, false);
    }
  }
#pragma unroll
  for (int nt = 0; nt < 4; ++nt) {
    int col = nt * 16 + m;
#pragma unroll
    for (int r = 0; r < 8; ++r) {
      int orow = strip * 16 + hlf * 8 + r;
      out[(size_t)orow * 64 + col] = frelu(acc[nt][r] + bl[col]);
    }
  }
}

// ---------------------------------------------------------------------------
// WMMA GEMM 2: con2 = relu([agg_c|con_feats|struc2](N x320) @ [Wc;Uc;Vc](320x128) + bc)
// N=128 processed in two 64-col phases (LDS 64x320 bf16 = 40KB); A frags cached.
// ---------------------------------------------------------------------------
__global__ __launch_bounds__(256) void k_con2(
    const float* __restrict__ aggc, const float* __restrict__ cf,
    const float* __restrict__ s2, const float* __restrict__ Wc,
    const float* __restrict__ Uc, const float* __restrict__ Vc,
    const float* __restrict__ bc, float* __restrict__ out) {
  __shared__ unsigned short wl[64 * 320];   // [col][k]
  __shared__ float bl[128];
  int wave = threadIdx.x >> 5, lane = threadIdx.x & 31;
  int strip = blockIdx.x * 8 + wave;
  bool active = strip < N_NODES / 16;
  int m = lane & 15, hlf = lane >> 4;
  int row = active ? strip * 16 + m : m;
  if (threadIdx.x < 128) bl[threadIdx.x] = bc[threadIdx.x];

  v16us af[10];
#pragma unroll
  for (int kc = 0; kc < 10; ++kc) {
    int kk0 = kc * 32 + hlf * 16, kk1 = kc * 32 + hlf * 16 + 8;
    const float* p0 = kk0 < 128 ? aggc + (size_t)row * 128 + kk0
                     : (kk0 < 256 ? cf + (size_t)row * 128 + (kk0 - 128)
                                  : s2 + (size_t)row * 64 + (kk0 - 256));
    const float* p1 = kk1 < 128 ? aggc + (size_t)row * 128 + kk1
                     : (kk1 < 256 ? cf + (size_t)row * 128 + (kk1 - 128)
                                  : s2 + (size_t)row * 64 + (kk1 - 256));
#pragma unroll
    for (int e = 0; e < 8; ++e) { af[kc][e] = f2bf(p0[e]); af[kc][8 + e] = f2bf(p1[e]); }
  }

  v8f zero;
#pragma unroll
  for (int e = 0; e < 8; ++e) zero[e] = 0.f;

  for (int ph = 0; ph < 2; ++ph) {
    __syncthreads();
    for (int t = (int)threadIdx.x; t < 320 * 64; t += 256) {
      int k = t >> 6, c = t & 63;
      int cc = c + ph * 64;
      float w = k < 128 ? Wc[k * 128 + cc]
               : (k < 256 ? Uc[(k - 128) * 128 + cc] : Vc[(k - 256) * 128 + cc]);
      wl[c * 320 + k] = f2bf(w);
    }
    __syncthreads();
    if (!active) continue;

    v8f acc[4];
#pragma unroll
    for (int t = 0; t < 4; ++t) acc[t] = zero;
#pragma unroll
    for (int kc = 0; kc < 10; ++kc) {
      int kkb = kc * 32 + hlf * 16;
#pragma unroll
      for (int nt = 0; nt < 4; ++nt) {
        int col = nt * 16 + m;
        v16us b = *reinterpret_cast<const v16us*>(wl + (size_t)col * 320 + kkb);
        acc[nt] = __builtin_amdgcn_wmma_f32_16x16x32_bf16(
            false, as_bf(af[kc]), false, as_bf(b), (short)0, acc[nt], false, false);
      }
    }
#pragma unroll
    for (int nt = 0; nt < 4; ++nt) {
      int col = ph * 64 + nt * 16 + m;
#pragma unroll
      for (int r = 0; r < 8; ++r) {
        int orow = strip * 16 + hlf * 8 + r;
        out[(size_t)orow * 128 + col] = frelu(acc[nt][r] + bl[col]);
      }
    }
  }
}

// ---------------------------------------------------------------------------
// WMMA GEMM 3 (edge MLP, fused): h = relu(concat(con2[src],con2[dst]) @ W1 + b1)
// esc = relu(h @ w2 + b2).  16 edges/wave, K=256, H=128 in two 64-col phases.
// ---------------------------------------------------------------------------
__global__ __launch_bounds__(256) void k_mlp(
    const float* __restrict__ con2, const int* __restrict__ eid,
    const float* __restrict__ W1, const float* __restrict__ b1,
    const float* __restrict__ w2, const float* __restrict__ b2,
    float* __restrict__ esc) {
  __shared__ unsigned short wl[64 * 256];   // [col][k]
  __shared__ float bl[128], w2l[128];
  int wave = threadIdx.x >> 5, lane = threadIdx.x & 31;
  int strip = blockIdx.x * 8 + wave;
  bool active = strip < E_PAIRS / 16;
  int m = lane & 15, hlf = lane >> 4;
  int erow = active ? strip * 16 + m : m;
  int ps = eid[erow], pd = eid[ETOT + erow];
  if (threadIdx.x < 128) { bl[threadIdx.x] = b1[threadIdx.x]; w2l[threadIdx.x] = w2[threadIdx.x]; }
  float b2v = b2[0];
  __builtin_prefetch(con2 + (size_t)ps * 128, 0, 0);
  __builtin_prefetch(con2 + (size_t)pd * 128, 0, 0);

  v16us af[8];
#pragma unroll
  for (int kc = 0; kc < 8; ++kc) {
    int kk0 = kc * 32 + hlf * 16, kk1 = kc * 32 + hlf * 16 + 8;
    const float* p0 = kk0 < 128 ? con2 + (size_t)ps * 128 + kk0
                                : con2 + (size_t)pd * 128 + (kk0 - 128);
    const float* p1 = kk1 < 128 ? con2 + (size_t)ps * 128 + kk1
                                : con2 + (size_t)pd * 128 + (kk1 - 128);
#pragma unroll
    for (int e = 0; e < 8; ++e) { af[kc][e] = f2bf(p0[e]); af[kc][8 + e] = f2bf(p1[e]); }
  }

  v8f zero;
#pragma unroll
  for (int e = 0; e < 8; ++e) zero[e] = 0.f;
  float part[8];
#pragma unroll
  for (int r = 0; r < 8; ++r) part[r] = 0.f;

  for (int ph = 0; ph < 2; ++ph) {
    __syncthreads();
    for (int t = (int)threadIdx.x; t < 256 * 64; t += 256) {
      int k = t >> 6, c = t & 63;
      wl[c * 256 + k] = f2bf(W1[k * 128 + c + ph * 64]);
    }
    __syncthreads();
    if (!active) continue;

    v8f acc[4];
#pragma unroll
    for (int t = 0; t < 4; ++t) acc[t] = zero;
#pragma unroll
    for (int kc = 0; kc < 8; ++kc) {
      int kkb = kc * 32 + hlf * 16;
#pragma unroll
      for (int nt = 0; nt < 4; ++nt) {
        int col = nt * 16 + m;
        v16us b = *reinterpret_cast<const v16us*>(wl + (size_t)col * 256 + kkb);
        acc[nt] = __builtin_amdgcn_wmma_f32_16x16x32_bf16(
            false, as_bf(af[kc]), false, as_bf(b), (short)0, acc[nt], false, false);
      }
    }
#pragma unroll
    for (int nt = 0; nt < 4; ++nt) {
      int col = ph * 64 + nt * 16 + m;
      float wv = w2l[col], bb = bl[col];
#pragma unroll
      for (int r = 0; r < 8; ++r) part[r] += frelu(acc[nt][r] + bb) * wv;
    }
  }
  if (!active) return;
  // reduce h@w2 across the 16 lanes holding this row's 128 columns
#pragma unroll
  for (int r = 0; r < 8; ++r) {
    float v = part[r];
    v += __shfl_xor(v, 8, 32);
    v += __shfl_xor(v, 4, 32);
    v += __shfl_xor(v, 2, 32);
    v += __shfl_xor(v, 1, 32);
    if (m == 0) esc[strip * 16 + hlf * 8 + r] = frelu(v + b2v);
  }
}

// ---------------------------------------------------------------------------
// Graph plumbing kernels
// ---------------------------------------------------------------------------
__global__ void k_build1(const int* __restrict__ eid, const float* __restrict__ esc,
                         unsigned* __restrict__ key, float* __restrict__ val) {
  int i = blockIdx.x * blockDim.x + threadIdx.x;
  if (i >= ETOT) return;
  key[i] = (unsigned)eid[i] + NN * (unsigned)eid[ETOT + i];
  float s = 1.f;
  if (i < 2 * E_PAIRS) s = esc[i < E_PAIRS ? i : i - E_PAIRS];
  val[i] = s;
}

__global__ void k_bitonic(unsigned* __restrict__ key, float* __restrict__ val, int j, int k) {
  int i = blockIdx.x * blockDim.x + threadIdx.x;
  int ixj = i ^ j;
  if (ixj <= i) return;
  unsigned ka = key[i], kb = key[ixj];
  bool up = (i & k) == 0;
  if ((ka > kb) == up) {
    key[i] = kb; key[ixj] = ka;
    float t = val[i]; val[i] = val[ixj]; val[ixj] = t;
  }
}

__global__ void k_denom(const unsigned* key, const float* val, float* denom) {
  int i = blockIdx.x * blockDim.x + threadIdx.x;
  if (i >= ETOT) return;
  atomicAdd(&denom[key[i] / NN], val[i]);
}
__global__ void k_norm(const unsigned* key, float* val, const float* denom) {
  int i = blockIdx.x * blockDim.x + threadIdx.x;
  if (i >= ETOT) return;
  val[i] /= denom[key[i] / NN];
}
__global__ void k_fillf(float* p, float v, int n) {
  int i = blockIdx.x * blockDim.x + threadIdx.x;
  if (i < n) p[i] = v;
}
__global__ void k_iota(int* p, int n) {
  int i = blockIdx.x * blockDim.x + threadIdx.x;
  if (i < n) p[i] = i;
}
__global__ void k_spmm(const unsigned* key, const float* val, const float* cur, float* nxt) {
  int i = blockIdx.x * blockDim.x + threadIdx.x;
  if (i >= ETOT) return;
  atomicAdd(&nxt[key[i] % NN], val[i] * cur[key[i] / NN]);
}
__global__ void k_cc(const unsigned* key, const float* val, const int* cur, int* nxt) {
  int i = blockIdx.x * blockDim.x + threadIdx.x;
  if (i >= ETOT) return;
  if (val[i] > 0.f) atomicMin(&nxt[key[i] / NN], cur[key[i] % NN]);
}
__global__ void k_present(const int* g, int* pres) {
  int i = blockIdx.x * blockDim.x + threadIdx.x;
  if (i < N_NODES) pres[g[i]] = 1;
}
// single-workgroup chunked inclusive scan
__global__ void k_scan(const int* __restrict__ in, int* __restrict__ out, int n) {
  __shared__ int buf[1024];
  __shared__ int carry;
  if (threadIdx.x == 0) carry = 0;
  __syncthreads();
  for (int base = 0; base < n; base += 1024) {
    int i = base + (int)threadIdx.x;
    buf[threadIdx.x] = (i < n) ? in[i] : 0;
    __syncthreads();
    for (int off = 1; off < 1024; off <<= 1) {
      int t = ((int)threadIdx.x >= off) ? buf[threadIdx.x - off] : 0;
      __syncthreads();
      buf[threadIdx.x] += t;
      __syncthreads();
    }
    if (i < n) out[i] = buf[threadIdx.x] + carry;
    __syncthreads();
    if (threadIdx.x == 0) carry += buf[1023];
    __syncthreads();
  }
}
__global__ void k_gidc(const int* g, const int* pcum, int* gidc) {
  int i = blockIdx.x * blockDim.x + threadIdx.x;
  if (i < N_NODES) gidc[i] = pcum[g[i]] - 1;
}
__global__ void k_scalar_ng(const int* pcum, int* ngdev, int* out_ng) {
  int v = pcum[N_NODES - 1]; ngdev[0] = v; out_ng[0] = v;
}
__global__ void k_gsize(const int* gidc, float* gs) {
  int i = blockIdx.x * blockDim.x + threadIdx.x;
  if (i < N_NODES) atomicAdd(&gs[gidc[i]], 1.f);
}
__global__ void k_nwdiv(float* nw, const int* gidc, const float* gs) {
  int i = blockIdx.x * blockDim.x + threadIdx.x;
  if (i < N_NODES) nw[i] /= gs[gidc[i]];
}
__global__ void k_pool(const float* feat, const float* nw, const int* gidc,
                       float* outp, int D, int chunks) {
  int idx = blockIdx.x * blockDim.x + threadIdx.x;
  if (idx >= N_NODES * chunks) return;
  int node = idx / chunks, f = (idx % chunks) * 4;
  float s = nw[node];
  int gr = gidc[node];
  const float* sp = feat + (size_t)node * D + f;
  float* dp = outp + (size_t)gr * D + f;
  atomicAdd(dp + 0, s * sp[0]); atomicAdd(dp + 1, s * sp[1]);
  atomicAdd(dp + 2, s * sp[2]); atomicAdd(dp + 3, s * sp[3]);
}
__global__ void k_edges2(const unsigned* key1, const float* eww, const float* nw,
                         const int* gidc, unsigned* key2, float* val2) {
  int i = blockIdx.x * blockDim.x + threadIdx.x;
  if (i >= ETOT) return;
  unsigned k = key1[i];
  int es = (int)(k % NN), ed = (int)(k / NN);
  key2[i] = (unsigned)gidc[es] * NN + (unsigned)gidc[ed];
  val2[i] = nw[es] * eww[i];
}
__global__ void k_isnew(const unsigned* key2, int* isnew) {
  int i = blockIdx.x * blockDim.x + threadIdx.x;
  if (i >= ETOT) return;
  isnew[i] = (i == 0) || (key2[i] != key2[i - 1]);
}
__global__ void k_coalesce(const unsigned* key2, const float* val2, const int* segin,
                           float* oew, int* osrc, int* odst) {
  int i = blockIdx.x * blockDim.x + threadIdx.x;
  if (i >= ETOT) return;
  int s = segin[i] - 1;
  atomicAdd(&oew[s], val2[i]);
  unsigned k = key2[i];
  osrc[s] = (int)(k / NN);
  odst[s] = (int)(k % NN);
}
__global__ void k_scalar_noe(const int* segin, int* out_noe) { out_noe[0] = segin[ETOT - 1]; }
__global__ void k_nbmin(const int* gidc, const int* nbi, int* nb) {
  int i = blockIdx.x * blockDim.x + threadIdx.x;
  if (i < N_NODES) atomicMin(&nb[gidc[i]], nbi[i]);
}
__global__ void k_nbfix(int* nb, const int* ngdev) {
  int i = blockIdx.x * blockDim.x + threadIdx.x;
  if (i < N_NODES && i >= ngdev[0]) nb[i] = 0;
}

// ---------------------------------------------------------------------------
extern "C" void kernel_launch(void* const* d_in, const int* in_sizes, int n_in,
                              void* d_out, int out_size, void* d_ws, size_t ws_size,
                              hipStream_t stream) {
  (void)in_sizes; (void)n_in; (void)out_size; (void)ws_size;
  const float* con_feats   = (const float*)d_in[0];
  const float* struc_feats = (const float*)d_in[1];
  const int*   edge_ids    = (const int*)d_in[2];
  const float* edge_w      = (const float*)d_in[3];
  const int*   node_batch  = (const int*)d_in[4];
  const float* Ws = (const float*)d_in[5];
  const float* Us = (const float*)d_in[6];
  const float* bs = (const float*)d_in[7];
  const float* Wc = (const float*)d_in[8];
  const float* Uc = (const float*)d_in[9];
  const float* Vc = (const float*)d_in[10];
  const float* bc = (const float*)d_in[11];
  const float* W1 = (const float*)d_in[12];
  const float* b1 = (const float*)d_in[13];
  const float* w2 = (const float*)d_in[14];
  const float* b2 = (const float*)d_in[15];

  // workspace layout (regions A/B are reused after the GEMM phase)
  char* wsp = (char*)d_ws;
  auto take = [&](size_t b) -> char* {
    char* p = wsp; wsp += (b + 255) & ~(size_t)255; return p;
  };
  char*  regionA = take((size_t)N_NODES * DSI * 4);   // agg_s -> sort1 + small arrays
  char*  regionB = take((size_t)N_NODES * DCI * 4);   // agg_c -> sort2 + flags
  float* struc2  = (float*)take((size_t)N_NODES * DSI * 4);
  float* con2    = (float*)take((size_t)N_NODES * DCI * 4);
  float* escbuf  = (float*)take((size_t)E_PAIRS * 4);

  float*    agg_s = (float*)regionA;
  float*    agg_c = (float*)regionB;
  unsigned* key1  = (unsigned*)regionA;
  float*    val1  = (float*)(regionA + (size_t)SORT_SZ * 4);
  char* pa = regionA + (size_t)SORT_SZ * 8;
  auto takeA = [&](size_t b) -> char* {
    char* p = pa; pa += (b + 255) & ~(size_t)255; return p;
  };
  float* denom   = (float*)takeA((size_t)N_NODES * 4);
  float* nwbuf0  = (float*)takeA((size_t)N_NODES * 4);
  float* nwbuf1  = (float*)takeA((size_t)N_NODES * 4);
  int*   g0      = (int*)takeA((size_t)N_NODES * 4);
  int*   g1      = (int*)takeA((size_t)N_NODES * 4);
  int*   present = (int*)takeA((size_t)N_NODES * 4);
  int*   pcum    = (int*)takeA((size_t)N_NODES * 4);
  int*   gidc    = (int*)takeA((size_t)N_NODES * 4);
  float* gsize   = (float*)takeA((size_t)N_NODES * 4);
  int*   ngdev   = (int*)takeA(256);

  unsigned* key2 = (unsigned*)regionB;
  float*    val2 = (float*)(regionB + (size_t)SORT_SZ * 4);
  int*      isnw = (int*)(regionB + (size_t)SORT_SZ * 8);
  int*      segi = (int*)(regionB + (size_t)SORT_SZ * 8 + (size_t)(ETOT + 64) * 4);

  // output layout (flat, return order)
  float* outf      = (float*)d_out;
  int*   outi      = (int*)d_out;
  float* con_out   = outf;
  float* struc_out = outf + (size_t)N_NODES * DCI;
  int*   out_src   = outi + (size_t)N_NODES * (DCI + DSI);
  int*   out_dst   = out_src + ETOT;
  float* out_ew    = (float*)(out_dst + ETOT);
  int*   out_nb    = (int*)(out_ew + ETOT);
  int*   out_gidc  = out_nb + N_NODES;
  int*   out_ng    = out_gidc + N_NODES;
  int*   out_noe   = out_ng + 1;

  const int TB = 256;
  const int EB = (ETOT + TB - 1) / TB;       // edge-sized grids
  const int NB = (N_NODES + TB - 1) / TB;    // node-sized grids

  // 1) edge aggregation
  hipMemsetAsync(agg_s, 0, (size_t)N_NODES * DSI * 4, stream);
  hipMemsetAsync(agg_c, 0, (size_t)N_NODES * DCI * 4, stream);
  k_agg<<<(ETOT * 16 + TB - 1) / TB, TB, 0, stream>>>(struc_feats, edge_ids, edge_w, agg_s, 64, 16);
  k_agg<<<(ETOT * 32 + TB - 1) / TB, TB, 0, stream>>>(con_feats, edge_ids, edge_w, agg_c, 128, 32);

  // 2) WMMA GEMMs
  k_struc2<<<(N_NODES / 16 + 7) / 8, TB, 0, stream>>>(agg_s, struc_feats, Ws, Us, bs, struc2);
  k_con2<<<(N_NODES / 16 + 7) / 8, TB, 0, stream>>>(agg_c, con_feats, struc2, Wc, Uc, Vc, bc, con2);
  k_mlp<<<(E_PAIRS / 16 + 7) / 8, TB, 0, stream>>>(con2, edge_ids, W1, b1, w2, b2, escbuf);

  // 3) build + sort directed edge list by key1 = src + n*dst
  hipMemsetAsync(key1, 0xFF, (size_t)SORT_SZ * 4, stream);
  k_build1<<<EB, TB, 0, stream>>>(edge_ids, escbuf, key1, val1);
  for (int k = 2; k <= SORT_SZ; k <<= 1)
    for (int j = k >> 1; j >= 1; j >>= 1)
      k_bitonic<<<SORT_SZ / TB, TB, 0, stream>>>(key1, val1, j, k);

  // 4) column-normalize scores
  hipMemsetAsync(denom, 0, (size_t)N_NODES * 4, stream);
  k_denom<<<EB, TB, 0, stream>>>(key1, val1, denom);
  k_norm<<<EB, TB, 0, stream>>>(key1, val1, denom);

  // 5) node-weight SpMM iterations
  float* nwp[2] = {nwbuf0, nwbuf1};
  k_fillf<<<NB, TB, 0, stream>>>(nwbuf0, 1.0f, N_NODES);
  for (int it = 0; it < 5; ++it) {
    float* cur = nwp[it & 1];
    float* nxt = nwp[(it + 1) & 1];
    hipMemsetAsync(nxt, 0, (size_t)N_NODES * 4, stream);
    k_spmm<<<EB, TB, 0, stream>>>(key1, val1, cur, nxt);
  }
  float* nwfin = nwp[1];

  // 6) connected components (fixed 100 label-propagation rounds; idempotent at fixpoint)
  int* gp[2] = {g0, g1};
  k_iota<<<NB, TB, 0, stream>>>(g0, N_NODES);
  for (int it = 0; it < 100; ++it) {
    int* cur = gp[it & 1];
    int* nxt = gp[(it + 1) & 1];
    hipMemsetAsync(nxt, 0x7F, (size_t)N_NODES * 4, stream);
    k_cc<<<EB, TB, 0, stream>>>(key1, val1, cur, nxt);
  }
  int* gfin = gp[0];

  // 7) compact group ids, sizes, weight division
  hipMemsetAsync(present, 0, (size_t)N_NODES * 4, stream);
  k_present<<<NB, TB, 0, stream>>>(gfin, present);
  k_scan<<<1, 1024, 0, stream>>>(present, pcum, N_NODES);
  k_gidc<<<NB, TB, 0, stream>>>(gfin, pcum, gidc);
  k_scalar_ng<<<1, 1, 0, stream>>>(pcum, ngdev, out_ng);
  hipMemsetAsync(gsize, 0, (size_t)N_NODES * 4, stream);
  k_gsize<<<NB, TB, 0, stream>>>(gidc, gsize);
  k_nwdiv<<<NB, TB, 0, stream>>>(nwfin, gidc, gsize);

  // 8) weighted pooling into groups
  hipMemsetAsync(con_out, 0, (size_t)N_NODES * DCI * 4, stream);
  hipMemsetAsync(struc_out, 0, (size_t)N_NODES * DSI * 4, stream);
  k_pool<<<(N_NODES * 32 + TB - 1) / TB, TB, 0, stream>>>(con2, nwfin, gidc, con_out, 128, 32);
  k_pool<<<(N_NODES * 16 + TB - 1) / TB, TB, 0, stream>>>(struc2, nwfin, gidc, struc_out, 64, 16);

  // 9) group edges + second sort
  hipMemsetAsync(key2, 0xFF, (size_t)SORT_SZ * 4, stream);
  k_edges2<<<EB, TB, 0, stream>>>(key1, edge_w, nwfin, gidc, key2, val2);
  for (int k = 2; k <= SORT_SZ; k <<= 1)
    for (int j = k >> 1; j >= 1; j >>= 1)
      k_bitonic<<<SORT_SZ / TB, TB, 0, stream>>>(key2, val2, j, k);

  // 10) coalesce duplicate group edges
  k_isnew<<<EB, TB, 0, stream>>>(key2, isnw);
  k_scan<<<1, 1024, 0, stream>>>(isnw, segi, ETOT);
  hipMemsetAsync(out_ew, 0, (size_t)ETOT * 4, stream);
  hipMemsetAsync(out_src, 0, (size_t)ETOT * 4, stream);
  hipMemsetAsync(out_dst, 0, (size_t)ETOT * 4, stream);
  k_coalesce<<<EB, TB, 0, stream>>>(key2, val2, segi, out_ew, out_src, out_dst);
  k_scalar_noe<<<1, 1, 0, stream>>>(segi, out_noe);

  // 11) group batch ids + gid_c output
  hipMemsetAsync(out_nb, 0x7F, (size_t)N_NODES * 4, stream);
  k_nbmin<<<NB, TB, 0, stream>>>(gidc, node_batch, out_nb);
  k_nbfix<<<NB, TB, 0, stream>>>(out_nb, ngdev);
  hipMemcpyAsync(out_gidc, gidc, (size_t)N_NODES * 4, hipMemcpyDeviceToDevice, stream);
}
